// Head_10144712753553
// MI455X (gfx1250) — matrix operand order repaired and checked by reading, not compiled, not run on hardware
//
#include <hip/hip_runtime.h>

// ---------------------------------------------------------------------------
// byte-scale attention for [B=8192, T=64, C=64, H=64] on gfx1250 (CDNA5)
// bf16 WMMA 16x16x32, f32 accumulate. One workgroup (4 waves) per batch item.
// Async global->LDS ingest when the toolchain exposes the gfx1250 builtins.
// ---------------------------------------------------------------------------

typedef __attribute__((ext_vector_type(4)))  __bf16 v4bf;
typedef __attribute__((ext_vector_type(8)))  __bf16 v8bf;
typedef __attribute__((ext_vector_type(16))) __bf16 v16bf;
typedef __attribute__((ext_vector_type(8)))  float  v8f;
typedef __attribute__((ext_vector_type(4)))  float  v4f;
typedef __attribute__((ext_vector_type(4)))  int    v4i;

typedef __attribute__((address_space(1))) v4i* gv4i_p;   // global <4 x i32>*
typedef __attribute__((address_space(3))) v4i* lv4i_p;   // LDS    <4 x i32>*

#define SB 72   // bf16 LDS row pitch (144B) -> conflict-free b128 row loads
#define SO 68   // f32 LDS row pitch for output staging

#if defined(__has_builtin)
#if __has_builtin(__builtin_amdgcn_global_load_async_to_lds_b128)
#define HAVE_ASYNC_LDS 1
#endif
#endif
#ifndef HAVE_ASYNC_LDS
#define HAVE_ASYNC_LDS 0
#endif

__device__ __forceinline__ void wait_asynccnt0() {
#if defined(__has_builtin)
#if __has_builtin(__builtin_amdgcn_s_wait_asynccnt)
  __builtin_amdgcn_s_wait_asynccnt(0);
  return;
#else
  asm volatile("s_wait_asynccnt 0x0" ::: "memory");
  return;
#endif
#else
  asm volatile("s_wait_asynccnt 0x0" ::: "memory");
#endif
}

// A fragment (16x32 bf16, row-major [M][K], pitch `stride` elements):
// lane L: M = L&15 ; elements 0..7 -> K = k0..k0+7, 8..15 -> K = k0+16..k0+23
// with k0 = kt*32 + (L>=16 ? 8 : 0)
__device__ __forceinline__ v16bf load_fragA(const __bf16* base, int mt, int kt,
                                            int lane, int stride) {
  int row = mt * 16 + (lane & 15);
  int k0  = kt * 32 + ((lane >> 4) << 3);
  const __bf16* p = base + row * stride + k0;
  v8bf lo = *(const v8bf*)(p);
  v8bf hi = *(const v8bf*)(p + 16);
  return __builtin_shufflevector(lo, hi, 0, 1, 2, 3, 4, 5, 6, 7,
                                 8, 9, 10, 11, 12, 13, 14, 15);
}

// B fragment (32x16 bf16) read from B^T stored row-major [N][K]:
// lane L: N = L&15 ; elements 0..7 -> K = k0..k0+7, 8..15 -> K = k0+16..k0+23, k0 = kt*32 + (L>=16?8:0)
__device__ __forceinline__ v16bf load_fragBT(const __bf16* baseT, int kt, int nt,
                                             int lane, int stride) {
  int row = nt * 16 + (lane & 15);
  int k0  = kt * 32 + ((lane >> 4) << 3);
  const __bf16* p = baseT + row * stride + k0;
  v8bf lo = *(const v8bf*)(p);
  v8bf hi = *(const v8bf*)(p + 16);
  return __builtin_shufflevector(lo, hi, 0, 1, 2, 3, 4, 5, 6, 7,
                                 8, 9, 10, 11, 12, 13, 14, 15);
}

__device__ __forceinline__ v8f wmma_bf16(v16bf a, v16bf b, v8f c) {
  return __builtin_amdgcn_wmma_f32_16x16x32_bf16(false, a, false, b,
                                                 (short)0, c, false, false);
}

__device__ __forceinline__ void cvt4_store(__bf16* dst, v4f d) {
  v4bf h;
  h.x = (__bf16)d.x; h.y = (__bf16)d.y; h.z = (__bf16)d.z; h.w = (__bf16)d.w;
  *(v4bf*)dst = h;
}

// Convert Wq/Wk/Wv (f32 [64][64], row = c_in, col = h_out) to bf16, stored
// TRANSPOSED [h_out][c_in] so B^T fragment loads are contiguous.
__global__ void cvt_weights_kernel(const float* __restrict__ Wq,
                                   const float* __restrict__ Wk,
                                   const float* __restrict__ Wv,
                                   __bf16* __restrict__ wT) {
  int idx = blockIdx.x * blockDim.x + threadIdx.x;   // 0 .. 3*4096-1
  if (idx >= 3 * 4096) return;
  int m = idx >> 12;
  int r = (idx >> 6) & 63;   // c_in
  int c = idx & 63;          // h_out
  const float* W = (m == 0) ? Wq : ((m == 1) ? Wk : Wv);
  wT[m * 4096 + c * 64 + r] = (__bf16)W[r * 64 + c];
}

__global__ __launch_bounds__(128) void head_attn_kernel(
    const float* __restrict__ x,      // [B,64,64] f32
    const __bf16* __restrict__ wT,    // 3 x [64][64] bf16 (transposed weights)
    float* __restrict__ out) {        // [B,64,64] f32
  __shared__ __align__(16) __bf16 smem[5 * 64 * SB];
  __bf16* xs = smem;                 // x, bf16, row-major [t][c]
  __bf16* ks = smem + 1 * 64 * SB;   // k,  row-major [s][h]
  __bf16* qs = smem + 2 * 64 * SB;   // q,  row-major [t][h]  (= B^T for scores)
  __bf16* vs = smem + 3 * 64 * SB;   // v^T, row-major [h][s]  (= B^T for out)
  __bf16* as = smem + 4 * 64 * SB;   // attn, row-major [t][s]
  float*  outs = (float*)smem;       // aliases xs+ks (dead in phase 3): 64*SO*4
  float*  xf32 = (float*)qs;         // f32 x staging, aliases qs..vs (16 KB)

  const int tid  = threadIdx.x;
  const int lane = tid & 31;
  const int wave = tid >> 5;
  const int b    = blockIdx.x;
  const float* xb = x + (size_t)b * 4096;

  // ---- phase 0: x -> bf16 LDS (b128 granularity) ------------------------
#if HAVE_ASYNC_LDS
#pragma unroll
  for (int p = 0; p < 8; ++p) {
    int j = (tid + p * 128) * 4;     // float index, 16B granules
    float* gsrc = const_cast<float*>(xb + j);
    __builtin_amdgcn_global_load_async_to_lds_b128(
        (gv4i_p)gsrc, (lv4i_p)(xf32 + j), 0, 0);
  }
  wait_asynccnt0();
  __syncthreads();
#pragma unroll
  for (int p = 0; p < 8; ++p) {
    int j = (tid + p * 128) * 4;
    v4f d = *(const v4f*)&xf32[j];
    int r = j >> 6, c = j & 63;
    cvt4_store(&xs[r * SB + c], d);
  }
#else
#pragma unroll
  for (int p = 0; p < 8; ++p) {
    int j = (tid + p * 128) * 4;
    v4f d = *(const v4f*)&xb[j];
    int r = j >> 6, c = j & 63;
    cvt4_store(&xs[r * SB + c], d);
  }
#endif
  __syncthreads();

  // ---- phase 1: q,k,v = x @ W (48 output tiles over 4 waves) -----------
#pragma unroll
  for (int m = 0; m < 3; ++m) {      // 0:q 1:k 2:v
    __bf16* dst = (m == 0) ? qs : ((m == 1) ? ks : vs);
#pragma unroll
    for (int i = 0; i < 4; ++i) {
      int t  = wave + 4 * i;         // tile index 0..15 within this matrix
      int mt = t >> 2, nt = t & 3;
      v8f acc = {0.f, 0.f, 0.f, 0.f, 0.f, 0.f, 0.f, 0.f};
#pragma unroll
      for (int kt = 0; kt < 2; ++kt) {
        v16bf A = load_fragA(xs, mt, kt, lane, SB);
        v16bf B = load_fragBT(wT + m * 4096, kt, nt, lane, 64);
        acc = wmma_bf16(A, B, acc);
      }
      int M0 = mt * 16 + ((lane >> 4) << 3);
      int N  = nt * 16 + (lane & 15);
#pragma unroll
      for (int r = 0; r < 8; ++r) {
        if (m == 2) dst[N * SB + (M0 + r)] = (__bf16)acc[r];
        else        dst[(M0 + r) * SB + N] = (__bf16)acc[r];
      }
    }
  }
  __syncthreads();

  // ---- phase 2: scores^T = k @ q^T ; mask + byte-scale -> attn ----------
  {
    const int nt = wave;                 // t-tile owned by this wave
    v8f acc[4];
#pragma unroll
    for (int mt = 0; mt < 4; ++mt) {
      acc[mt] = (v8f){0.f, 0.f, 0.f, 0.f, 0.f, 0.f, 0.f, 0.f};
#pragma unroll
      for (int kt = 0; kt < 2; ++kt) {
        v16bf A = load_fragA(ks, mt, kt, lane, SB);   // A = k  [s][h]
        v16bf B = load_fragBT(qs, kt, nt, lane, SB);  // B^T = q [t][h]
        acc[mt] = wmma_bf16(A, B, acc[mt]);
      }
    }
    const float scale = 0.125f;                 // 1/sqrt(64)
    const int tcol = nt * 16 + (lane & 15);     // global t (column of scores^T)
    const int M0b  = (lane >> 4) << 3;
    float mrow = 0.f;
#pragma unroll
    for (int mt = 0; mt < 4; ++mt) {
#pragma unroll
      for (int r = 0; r < 8; ++r) {
        int s = mt * 16 + M0b + r;              // global s (row of scores^T)
        float f = (s <= tcol) ? fmaxf(acc[mt][r] * scale, 0.f) : 0.f;
        acc[mt][r] = f;
        mrow = fmaxf(mrow, f);
      }
    }
    mrow = fmaxf(mrow, __shfl_xor(mrow, 16, 32));   // partner holds other M-half
    const float inv = 256.f / (mrow + 1e-6f);
#pragma unroll
    for (int mt = 0; mt < 4; ++mt) {
      v8bf pk;
#pragma unroll
      for (int r = 0; r < 8; ++r) {
        float w = floorf(acc[mt][r] * inv);
        w = fminf(w, 255.f);
        if (w > 127.f) w -= 256.f;              // int8 wrap, per reference
        pk[r] = (__bf16)(w * (1.f / 256.f));    // exact in bf16
      }
      // store transposed -> attn row-major [t][s]; 8 contiguous s per lane
      *(v8bf*)&as[tcol * SB + mt * 16 + M0b] = pk;
    }
  }
  __syncthreads();

  // ---- phase 3: out = attn @ v -----------------------------------------
  {
    const int mt = wave;                // t-tile row owned by this wave
#pragma unroll
    for (int nt = 0; nt < 4; ++nt) {
      v8f acc = {0.f, 0.f, 0.f, 0.f, 0.f, 0.f, 0.f, 0.f};
#pragma unroll
      for (int kt = 0; kt < 2; ++kt) {
        v16bf A = load_fragA(as, mt, kt, lane, SB);   // A = attn [t][s]
        v16bf B = load_fragBT(vs, kt, nt, lane, SB);  // B^T = v^T [h][s]
        acc = wmma_bf16(A, B, acc);
      }
      int M0 = mt * 16 + ((lane >> 4) << 3);
      int N  = nt * 16 + (lane & 15);
#pragma unroll
      for (int r = 0; r < 8; ++r) outs[(M0 + r) * SO + N] = acc[r];
    }
  }
  __syncthreads();

  // ---- coalesced float4 writeback --------------------------------------
  float* ob = out + (size_t)b * 4096;
#pragma unroll
  for (int p = 0; p < 8; ++p) {
    int j = tid + p * 128;          // float4 index 0..1023
    int flat = j * 4;
    int r = flat >> 6, c = flat & 63;
    *(v4f*)&ob[flat] = *(const v4f*)&outs[r * SO + c];
  }
}

extern "C" void kernel_launch(void* const* d_in, const int* in_sizes, int n_in,
                              void* d_out, int out_size, void* d_ws, size_t ws_size,
                              hipStream_t stream) {
  const float* x  = (const float*)d_in[0];
  const float* Wq = (const float*)d_in[1];
  const float* Wk = (const float*)d_in[2];
  const float* Wv = (const float*)d_in[3];
  float* out = (float*)d_out;
  __bf16* wT = (__bf16*)d_ws;   // 3*4096*2 = 24 KB scratch

  cvt_weights_kernel<<<(3 * 4096 + 255) / 256, 256, 0, stream>>>(Wq, Wk, Wv, wT);

  int batch = in_sizes[0] / 4096;   // 8192
  head_attn_kernel<<<batch, 128, 0, stream>>>(x, wT, out);
}
